// DecoderRNN_35923106464218
// MI455X (gfx1250) — compile-verified, matrix operand order checked by reading, not run on hardware
//
#include <hip/hip_runtime.h>
#include <math.h>

// ---------------------------------------------------------------------------
// DecoderRNN (teacher-forced GRU decoder) for MI455X / gfx1250.
//
// - All GEMMs on v_wmma_f32_16x16x32_bf16. Weights AND the embedding table
//   are converted once to bf16 in workspace (W_out bf16 = 65.5MB + emb bf16 =
//   32.8MB both L2-resident on the 192MB L2), so the per-step hot loops are
//   pure global_load_b128 -> v_wmma with no conversion VALU.
// - Per-step critical path = 2 kernels (GRU with fused bf16 embedding gather;
//   logits GEMM with fused per-block softmax partials). All per-(t,b)
//   bookkeeping (argmax, EOS mask chain, NLL) runs in ONE end kernel (mask
//   recurrence is independent across batch rows), then one wide kernel
//   applies the log-softmax normalization in a single pass.
// - 136 kernel launches per call, all graph-capturable on `stream`.
//
// Workspace (~112 MB):
//   Wih_bf [3D,E] | Whh_bf [3D,D] | Wout_bf [V,D] | emb_bf [V,E] (bf16) |
//   Hbf x2 [B,D] bf16 | Hf x2 [B,D] f32 | nll [T,B] f32 | logZ [T,B] f32 |
//   partials [T,125,B] float4
// ---------------------------------------------------------------------------

typedef __bf16 bf16;
typedef __attribute__((ext_vector_type(16))) __bf16 v16bf;
typedef __attribute__((ext_vector_type(8)))  float  v8f;

static constexpr int B = 32, T = 64, E = 512, D = 1024, ENC = 1024, V = 32000;
static constexpr int PAD_IDX = 0, EOS_IDX = 2;
static constexpr int NCOL = 256;           // logit columns per block
static constexpr int NBLK = V / NCOL;      // 125 logit blocks

// d_out layout: log_probs [T,B,V] | preds [B,T] | mask [T,B] | loss [1]
static constexpr size_t PREDS_OFF = (size_t)T * B * V;
static constexpr size_t MASK_OFF  = PREDS_OFF + (size_t)B * T;
static constexpr size_t LOSS_OFF  = MASK_OFF + (size_t)T * B;

__device__ __forceinline__ bf16 f2bf(float f) {
  union { float f; unsigned u; } a; a.f = f;
  unsigned r = a.u + 0x7fffu + ((a.u >> 16) & 1u);   // round-to-nearest-even
  unsigned short h = (unsigned short)(r >> 16);
  bf16 b; __builtin_memcpy(&b, &h, 2);
  return b;
}

// 16x32 bf16 WMMA fragment. ISA 16-bit layout: lanes 0-15 hold row (lane),
// elems 0..7 = K+0..7, elems 8..15 = K+16..23; lanes 16-31 the +8/+24 halves.
__device__ __forceinline__ v16bf load_frag(const bf16* base, int ld, int lane) {
  const bf16* p = base + (size_t)(lane & 15) * ld + ((lane >> 4) << 3);
  v16bf v;
#pragma unroll
  for (int i = 0; i < 8; ++i) v[i] = p[i];
#pragma unroll
  for (int i = 0; i < 8; ++i) v[8 + i] = p[16 + i];
  return v;
}

// Fragment gather from a per-lane bf16 row pointer (embedding row).
__device__ __forceinline__ v16bf load_frag_row(const bf16* rowp, int k,
                                               int lane) {
  const bf16* p = rowp + k + ((lane >> 4) << 3);
  v16bf v;
#pragma unroll
  for (int i = 0; i < 8; ++i) v[i] = p[i];
#pragma unroll
  for (int i = 0; i < 8; ++i) v[8 + i] = p[16 + i];
  return v;
}

__device__ __forceinline__ v8f wmma_bf16(v16bf a, v16bf b, v8f c) {
  return __builtin_amdgcn_wmma_f32_16x16x32_bf16(
      false, a, false, b, (short)0, c, false, false);
}

// --------------------------- one-time kernels ------------------------------

__global__ void cvt_bf16_kernel(const float* __restrict__ src,
                                bf16* __restrict__ dst, int n) {
  for (int i = blockIdx.x * blockDim.x + threadIdx.x; i < n;
       i += gridDim.x * blockDim.x)
    dst[i] = f2bf(src[i]);
}

__global__ void init_h0_kernel(const float* __restrict__ ef,
                               const float* __restrict__ W_enc,
                               const float* __restrict__ b_enc,
                               float* __restrict__ Hf,
                               bf16* __restrict__ Hbf) {
  int idx = blockIdx.x * blockDim.x + threadIdx.x;   // B*D threads
  int b = idx >> 10, d = idx & (D - 1);
  const float* x = ef + (size_t)b * ENC;             // encoder_final[0][b]
  const float* w = W_enc + (size_t)d * ENC;
  float acc = b_enc[d];
  for (int k = 0; k < ENC; ++k) acc = fmaf(x[k], w[k], acc);
  float h = tanhf(acc);
  Hf[idx] = h;
  Hbf[idx] = f2bf(h);
}

// --------------------------- per-step kernels ------------------------------

// One wave owns 16 gate columns (d0..d0+15) for all 3 gates and both M-tiles.
// r/z accumulate gi+gh together; n keeps gi/gh split (n = tanh(inn + r*hn)).
// x_t is gathered straight from the bf16 embedding table by token (fused
// embed); at t==0 x==0 so the X GEMM is skipped. 64 waves total (D/16).
__global__ void gru_step_kernel(const int* __restrict__ trg,
                                const bf16* __restrict__ emb_bf,
                                const bf16* __restrict__ Hbf,
                                const float* __restrict__ Hf,
                                const bf16* __restrict__ Wih,
                                const bf16* __restrict__ Whh,
                                const float* __restrict__ b_ih,
                                const float* __restrict__ b_hh,
                                float* __restrict__ Hf_n,
                                bf16* __restrict__ Hbf_n, int t) {
  int wave = (blockIdx.x * blockDim.x + threadIdx.x) >> 5;  // 0..63
  int lane = threadIdx.x & 31;
  int d0 = wave << 4;

  v8f aR0 = {}, aR1 = {}, aZ0 = {}, aZ1 = {};
  v8f aIN0 = {}, aIN1 = {}, aHN0 = {}, aHN1 = {};

  if (t > 0) {
    // embedding rows for this lane's A-fragment rows (rows = batch index)
    int r16 = lane & 15;
    int tok0 = trg[r16 * T + (t - 1)];
    int tok1 = trg[(16 + r16) * T + (t - 1)];
    const bf16* e0 = emb_bf + (size_t)tok0 * E;
    const bf16* e1 = emb_bf + (size_t)tok1 * E;
    for (int k = 0; k < E; k += 32) {
      v16bf x0 = load_frag_row(e0, k, lane);
      v16bf x1 = load_frag_row(e1, k, lane);
      v16bf br = load_frag(Wih + (size_t)(0 * D + d0) * E + k, E, lane);
      aR0 = wmma_bf16(x0, br, aR0);  aR1 = wmma_bf16(x1, br, aR1);
      v16bf bz = load_frag(Wih + (size_t)(1 * D + d0) * E + k, E, lane);
      aZ0 = wmma_bf16(x0, bz, aZ0);  aZ1 = wmma_bf16(x1, bz, aZ1);
      v16bf bn = load_frag(Wih + (size_t)(2 * D + d0) * E + k, E, lane);
      aIN0 = wmma_bf16(x0, bn, aIN0); aIN1 = wmma_bf16(x1, bn, aIN1);
    }
  }
  for (int k = 0; k < D; k += 32) {
    v16bf h0 = load_frag(Hbf + k, D, lane);
    v16bf h1 = load_frag(Hbf + (size_t)16 * D + k, D, lane);
    v16bf br = load_frag(Whh + (size_t)(0 * D + d0) * D + k, D, lane);
    aR0 = wmma_bf16(h0, br, aR0);  aR1 = wmma_bf16(h1, br, aR1);
    v16bf bz = load_frag(Whh + (size_t)(1 * D + d0) * D + k, D, lane);
    aZ0 = wmma_bf16(h0, bz, aZ0);  aZ1 = wmma_bf16(h1, bz, aZ1);
    v16bf bn = load_frag(Whh + (size_t)(2 * D + d0) * D + k, D, lane);
    aHN0 = wmma_bf16(h0, bn, aHN0); aHN1 = wmma_bf16(h1, bn, aHN1);
  }

  // C layout: VGPR j of lane l -> (m = (l>=16)*8 + j, n = l&15)
  int d  = d0 + (lane & 15);
  int rb = (lane >> 4) << 3;
  float br_  = b_ih[d] + b_hh[d];
  float bz_  = b_ih[D + d] + b_hh[D + d];
  float bin_ = b_ih[2 * D + d];
  float bhn_ = b_hh[2 * D + d];
#pragma unroll
  for (int j = 0; j < 8; ++j) {
    {
      int m = rb + j;
      float r = 1.f / (1.f + __expf(-(aR0[j] + br_)));
      float z = 1.f / (1.f + __expf(-(aZ0[j] + bz_)));
      float n = tanhf(aIN0[j] + bin_ + r * (aHN0[j] + bhn_));
      float h = (1.f - z) * n + z * Hf[m * D + d];
      Hf_n[m * D + d] = h;
      Hbf_n[m * D + d] = f2bf(h);
    }
    {
      int m = 16 + rb + j;
      float r = 1.f / (1.f + __expf(-(aR1[j] + br_)));
      float z = 1.f / (1.f + __expf(-(aZ1[j] + bz_)));
      float n = tanhf(aIN1[j] + bin_ + r * (aHN1[j] + bhn_));
      float h = (1.f - z) * n + z * Hf[m * D + d];
      Hf_n[m * D + d] = h;
      Hbf_n[m * D + d] = f2bf(h);
    }
  }
}

// logits[t] = H_new @ W_out^T, raw logits into the d_out log_probs region.
// 8 waves/block, each a 32x32 output tile (4 accumulators). H staged once per
// block in LDS (64KB); W_out rows stream from the L2-resident bf16 copy.
// After the GEMM the block emits per-row softmax partials (max, sum-exp,
// argmax) for its 256 columns, reusing the dead LDS staging buffer.
__global__ void logits_kernel(const bf16* __restrict__ Hbf,
                              const bf16* __restrict__ Wout,
                              float* __restrict__ out,
                              float4* __restrict__ partials, int t) {
  __shared__ bf16 sH[B * D];                        // 64 KB (reused for stats)
  {
    const uint4* s = (const uint4*)Hbf;
    uint4* dp = (uint4*)sH;
    for (int i = threadIdx.x; i < (B * D) / 8; i += blockDim.x) dp[i] = s[i];
  }
  __syncthreads();

  int wave = threadIdx.x >> 5, lane = threadIdx.x & 31;
  int n0 = blockIdx.x * NCOL + wave * 32;           // two 16-col tiles
  v8f c00 = {}, c01 = {}, c10 = {}, c11 = {};
  for (int k = 0; k < D; k += 32) {
    v16bf a0 = load_frag(sH + k, D, lane);
    v16bf a1 = load_frag(sH + (size_t)16 * D + k, D, lane);
    v16bf b0 = load_frag(Wout + (size_t)n0 * D + k, D, lane);
    v16bf b1 = load_frag(Wout + (size_t)(n0 + 16) * D + k, D, lane);
    c00 = wmma_bf16(a0, b0, c00);
    c10 = wmma_bf16(a1, b0, c10);
    c01 = wmma_bf16(a0, b1, c01);
    c11 = wmma_bf16(a1, b1, c11);
  }
  size_t base = (size_t)t * B * V;
  int col = lane & 15;
  int rb = (lane >> 4) << 3;
#pragma unroll
  for (int j = 0; j < 8; ++j) {
    out[base + (size_t)(rb + j) * V + n0 + col]           = c00[j];
    out[base + (size_t)(16 + rb + j) * V + n0 + col]      = c10[j];
    out[base + (size_t)(rb + j) * V + n0 + 16 + col]      = c01[j];
    out[base + (size_t)(16 + rb + j) * V + n0 + 16 + col] = c11[j];
  }

  __syncthreads();                 // all A-fragment reads of sH done
  float* sStat = (float*)sH;       // [32 rows][256 cols]
  int lc = wave * 32 + col;
#pragma unroll
  for (int j = 0; j < 8; ++j) {
    sStat[(rb + j) * NCOL + lc]           = c00[j];
    sStat[(16 + rb + j) * NCOL + lc]      = c10[j];
    sStat[(rb + j) * NCOL + lc + 16]      = c01[j];
    sStat[(16 + rb + j) * NCOL + lc + 16] = c11[j];
  }
  __syncthreads();
  if (threadIdx.x < 32) {
    int row = threadIdx.x;
    const float* p = sStat + row * NCOL;
    float m = p[0]; int bi = 0;
    for (int c = 1; c < NCOL; ++c)
      if (p[c] > m) { m = p[c]; bi = c; }            // first-index ties
    float s = 0.f;
    for (int c = 0; c < NCOL; ++c) s += __expf(p[c] - m);
    partials[(size_t)(t * NBLK + blockIdx.x) * 32 + row] =
        float4{m, s, (float)(blockIdx.x * NCOL + bi), 0.f};
  }
}

// --------------------------- end-of-sequence -------------------------------

// One block per batch row b; loops t sequentially (the EOS-mask recurrence is
// per-b). Combines the 125 softmax partials per (t,b), emits preds/mask/nll
// and logZ for the normalization pass. All reductions fixed-order.
__global__ void bookkeep_kernel(float* __restrict__ out,
                                const int* __restrict__ trg,
                                const float4* __restrict__ partials,
                                float* __restrict__ logZ_ws,
                                float* __restrict__ nll_ws) {
  int b = blockIdx.x;
  __shared__ float sm[256], ss[256], sbv[256];
  __shared__ int sbi[256];
  int stepm = 1;                                    // thread 0's copy matters

  for (int t = 0; t < T; ++t) {
    float m = -__builtin_inff(), s = 0.f, bv = -__builtin_inff();
    int bi = 0x7fffffff;
    if ((int)threadIdx.x < NBLK) {
      float4 p = partials[(size_t)(t * NBLK + threadIdx.x) * 32 + b];
      m = p.x; s = p.y; bv = p.x; bi = (int)p.z;
    }
    sm[threadIdx.x] = m; ss[threadIdx.x] = s;
    sbv[threadIdx.x] = bv; sbi[threadIdx.x] = bi;
    __syncthreads();
    for (int off = 128; off > 0; off >>= 1) {
      if ((int)threadIdx.x < off) {
        float m1 = sm[threadIdx.x], m2 = sm[threadIdx.x + off];
        float M = fmaxf(m1, m2);
        ss[threadIdx.x] = ss[threadIdx.x] * __expf(m1 - M) +
                          ss[threadIdx.x + off] * __expf(m2 - M);
        sm[threadIdx.x] = M;
        float v2 = sbv[threadIdx.x + off]; int i2 = sbi[threadIdx.x + off];
        if (v2 > sbv[threadIdx.x] ||
            (v2 == sbv[threadIdx.x] && i2 < sbi[threadIdx.x])) {
          sbv[threadIdx.x] = v2; sbi[threadIdx.x] = i2;
        }
      }
      __syncthreads();
    }
    if (threadIdx.x == 0) {
      float logZ = sm[0] + __logf(ss[0]);
      int pred = sbi[0];
      int tgt = trg[b * T + t];                      // trg_t[t][b]
      float nll = -(out[((size_t)t * B + b) * V + tgt] - logZ);
      nll_ws[t * B + b] = (stepm && tgt != PAD_IDX) ? nll : 0.f;
      out[PREDS_OFF + (size_t)b * T + t] = (float)pred;
      out[MASK_OFF + (size_t)t * B + b] = stepm ? 1.f : 0.f;
      logZ_ws[t * B + b] = logZ;
      stepm = (stepm && pred != EOS_IDX) ? 1 : 0;
    }
    __syncthreads();
  }
}

// log_probs[t][b][:] -= logZ[t][b], one block per (t,b) row.
__global__ void normalize_kernel(float* __restrict__ out,
                                 const float* __restrict__ logZ_ws) {
  int row = blockIdx.x;                              // t*B + b
  float lz = logZ_ws[row];
  float* p = out + (size_t)row * V;
  for (int v = threadIdx.x; v < V; v += blockDim.x) p[v] -= lz;
}

__global__ void loss_kernel(const float* __restrict__ nll_ws,
                            float* __restrict__ out) {
  __shared__ float red[256];
  float acc = 0.f;
  for (int i = threadIdx.x; i < T * B; i += 256) acc += nll_ws[i];
  red[threadIdx.x] = acc;
  __syncthreads();
  for (int off = 128; off > 0; off >>= 1) {
    if ((int)threadIdx.x < off) red[threadIdx.x] += red[threadIdx.x + off];
    __syncthreads();
  }
  if (threadIdx.x == 0) out[LOSS_OFF] = red[0] / (float)B;
}

// ------------------------------- launcher ----------------------------------

extern "C" void kernel_launch(void* const* d_in, const int* in_sizes, int n_in,
                              void* d_out, int out_size, void* d_ws,
                              size_t ws_size, hipStream_t stream) {
  const float* encoder_final = (const float*)d_in[0];
  // d_in[1] encoder_outputs: only defines batch size, unused
  const int*   trg   = (const int*)d_in[2];
  const float* emb   = (const float*)d_in[3];
  const float* W_ih  = (const float*)d_in[4];
  const float* W_hh  = (const float*)d_in[5];
  const float* b_ih  = (const float*)d_in[6];
  const float* b_hh  = (const float*)d_in[7];
  const float* W_enc = (const float*)d_in[8];
  const float* b_enc = (const float*)d_in[9];
  const float* W_out = (const float*)d_in[10];
  float* out = (float*)d_out;

  char* w = (char*)d_ws;
  bf16* Wih_bf  = (bf16*)w;  w += (size_t)3 * D * E * sizeof(bf16);
  bf16* Whh_bf  = (bf16*)w;  w += (size_t)3 * D * D * sizeof(bf16);
  bf16* Wout_bf = (bf16*)w;  w += (size_t)V * D * sizeof(bf16);
  bf16* emb_bf  = (bf16*)w;  w += (size_t)V * E * sizeof(bf16);
  bf16* Hbf0    = (bf16*)w;  w += (size_t)B * D * sizeof(bf16);
  bf16* Hbf1    = (bf16*)w;  w += (size_t)B * D * sizeof(bf16);
  float* Hf0    = (float*)w; w += (size_t)B * D * sizeof(float);
  float* Hf1    = (float*)w; w += (size_t)B * D * sizeof(float);
  float*  nll_ws   = (float*)w;  w += (size_t)T * B * sizeof(float);
  float*  logZ_ws  = (float*)w;  w += (size_t)T * B * sizeof(float);
  float4* partials = (float4*)w; // T * NBLK * 32 float4 = 4 MB

  bf16*  Hbf[2] = {Hbf0, Hbf1};
  float* Hf[2]  = {Hf0, Hf1};

  // one-time: bf16 weight + embedding copies, then h0
  cvt_bf16_kernel<<<2048, 256, 0, stream>>>(W_ih, Wih_bf, 3 * D * E);
  cvt_bf16_kernel<<<2048, 256, 0, stream>>>(W_hh, Whh_bf, 3 * D * D);
  cvt_bf16_kernel<<<4096, 256, 0, stream>>>(W_out, Wout_bf, V * D);
  cvt_bf16_kernel<<<4096, 256, 0, stream>>>(emb, emb_bf, V * E);
  init_h0_kernel<<<(B * D) / 256, 256, 0, stream>>>(encoder_final, W_enc,
                                                    b_enc, Hf[0], Hbf[0]);

  for (int t = 0; t < T; ++t) {
    int cur = t & 1, nxt = cur ^ 1;
    gru_step_kernel<<<16, 128, 0, stream>>>(trg, emb_bf, Hbf[cur], Hf[cur],
                                            Wih_bf, Whh_bf, b_ih, b_hh,
                                            Hf[nxt], Hbf[nxt], t);
    logits_kernel<<<NBLK, 256, 0, stream>>>(Hbf[nxt], Wout_bf, out, partials,
                                            t);
  }
  bookkeep_kernel<<<B, 256, 0, stream>>>(out, trg, partials, logZ_ws, nll_ws);
  normalize_kernel<<<T * B, 256, 0, stream>>>(out, logZ_ws);
  loss_kernel<<<1, 256, 0, stream>>>(nll_ws, out);
}
